// GNN_learner_58634893525795
// MI455X (gfx1250) — matrix operand order repaired and checked by reading, not compiled, not run on hardware
//
#include <hip/hip_runtime.h>
#include <hip/hip_bf16.h>

typedef __bf16 bf16;
typedef __attribute__((ext_vector_type(16))) __bf16 v16bf;
typedef __attribute__((ext_vector_type(8)))  float  v8f;
typedef int v4i __attribute__((vector_size(16)));   // matches builtin's pointee type

#ifndef __has_builtin
#define __has_builtin(x) 0
#endif

#if __has_builtin(__builtin_amdgcn_global_load_async_to_lds_b128)
#define HAVE_ASYNC_CP 1
#else
#define HAVE_ASYNC_CP 0
#endif

#if __has_builtin(__builtin_amdgcn_s_wait_asynccnt)
#define ASYNC_WAIT(n) __builtin_amdgcn_s_wait_asynccnt(n)
#elif HAVE_ASYNC_CP
#define ASYNC_WAIT(n) asm volatile("s_wait_asynccnt %0" ::"i"(n) : "memory")
#else
#define ASYNC_WAIT(n) ((void)0)
#endif

// 16-byte global -> LDS copy: async (CDNA5 path) or sync fallback.
__device__ __forceinline__ void cp16_g2l(const bf16* gsrc, bf16* ldst)
{
#if HAVE_ASYNC_CP
    __builtin_amdgcn_global_load_async_to_lds_b128(
        (__attribute__((address_space(1))) v4i*)gsrc,
        (__attribute__((address_space(3))) v4i*)ldst, 0, 0);
#else
    *(uint4*)ldst = *(const uint4*)gsrc;
#endif
}

#define BM 128
#define BN 128
#define BK 32
#define LDT 40   // LDS tile row stride in bf16 (32 + 8 pad): 80B, 16B aligned, bank-spread

union Frag16 { uint4 q[2]; v16bf v; };
union Pack8  { bf16 h[8]; uint4 q; };

// =====================================================================
// Primary GEMM: A[M][K] bf16, B[N][K] bf16 (both K-contiguous), async-to-LDS
// double-buffered staging, bf16 WMMA fp32 accumulate.
// mode: 0 -> fp32 C[M][N]; 1 -> bf16 relu C[M][N]; 2 -> bf16 C^T[N][M]
// =====================================================================
__global__ __launch_bounds__(256)
void gemm_nt_bf16_async(const bf16* __restrict__ A, const bf16* __restrict__ B,
                        void* __restrict__ Cout, int M, int N, int K, int mode)
{
    __shared__ bf16 As[2][BM * LDT];
    __shared__ bf16 Bs[2][BN * LDT];

    const int tid   = threadIdx.x;
    const int wid   = tid >> 5;
    const int lane  = tid & 31;
    const int lrow  = lane & 15;
    const int lhalf = lane >> 4;
    const int wm    = wid >> 1;     // 0..3 -> 32-row slab
    const int wn    = wid & 1;      // 0..1 -> 64-col slab
    const int bm0   = blockIdx.y * BM;
    const int bn0   = blockIdx.x * BN;

    v8f acc[2][4] = {};

    // Each thread copies 2 A-chunks + 2 B-chunks of 16B per tile (512 chunks each side).
    // A-side LDS placement is k-contiguous (same as B):
    // chunk c (k-group base c*8) -> LDS offset c*8.
    auto issue_tile = [&](int kt, int buf) {
#pragma unroll
        for (int i = 0; i < 2; i++) {
            const int id = tid + 256 * i;
            const int r = id >> 2, c = id & 3;
            const int pc = c * 8;
            cp16_g2l(&A[(size_t)(bm0 + r) * K + kt * BK + c * 8], &As[buf][r * LDT + pc]);
        }
#pragma unroll
        for (int i = 0; i < 2; i++) {
            const int id = tid + 256 * i;
            const int r = id >> 2, c = id & 3;
            cp16_g2l(&B[(size_t)(bn0 + r) * K + kt * BK + c * 8], &Bs[buf][r * LDT + c * 8]);
        }
    };

    const int nk = K / BK;
    issue_tile(0, 0);

    for (int kt = 0; kt < nk; kt++) {
        const int buf = kt & 1;
        if (kt + 1 < nk) {
            issue_tile(kt + 1, buf ^ 1);
            ASYNC_WAIT(4);            // newest 4 outstanding = next tile; tile kt done
        } else {
            ASYNC_WAIT(0);
        }
        __syncthreads();

        Frag16 afr[2], bfr[4];
        const int m0 = wm * 32;
        const int n0 = wn * 64;
#pragma unroll
        for (int i = 0; i < 2; i++) {
            const bf16* p = &As[buf][(m0 + i * 16 + lrow) * LDT + 16 * lhalf];
            afr[i].q[0] = *(const uint4*)p;
            afr[i].q[1] = *(const uint4*)(p + 8);
        }
#pragma unroll
        for (int j = 0; j < 4; j++) {
            const bf16* p = &Bs[buf][(n0 + j * 16 + lrow) * LDT + 16 * lhalf];
            bfr[j].q[0] = *(const uint4*)p;
            bfr[j].q[1] = *(const uint4*)(p + 8);
        }
#pragma unroll
        for (int i = 0; i < 2; i++)
#pragma unroll
            for (int j = 0; j < 4; j++)
                acc[i][j] = __builtin_amdgcn_wmma_f32_16x16x32_bf16(
                    false, afr[i].v, false, bfr[j].v, (short)0, acc[i][j], false, false);
        __syncthreads();
    }

    // Writeback. C/D layout: lane col = lrow, accum elem r -> row 8*lhalf + r.
#pragma unroll
    for (int i = 0; i < 2; i++)
#pragma unroll
        for (int j = 0; j < 4; j++) {
            const int gm = bm0 + wm * 32 + i * 16 + 8 * lhalf;
            const int gn = bn0 + wn * 64 + j * 16 + lrow;
            if (mode == 0) {
                float* Cf = (float*)Cout;
#pragma unroll
                for (int r = 0; r < 8; r++)
                    Cf[(size_t)(gm + r) * N + gn] = acc[i][j][r];
            } else if (mode == 1) {
                bf16* Ch = (bf16*)Cout;
#pragma unroll
                for (int r = 0; r < 8; r++)
                    Ch[(size_t)(gm + r) * N + gn] = (bf16)fmaxf(acc[i][j][r], 0.0f);
            } else {
                Pack8 p;
#pragma unroll
                for (int r = 0; r < 8; r++) p.h[r] = (bf16)acc[i][j][r];
                *(uint4*)&((bf16*)Cout)[(size_t)gn * M + gm] = p.q;  // C^T, 16B store
            }
        }
}

// =====================================================================
// Legacy fp32-operand GEMM (fallback when ws is too small for bf16 copies)
// =====================================================================
template <bool TB>
__global__ __launch_bounds__(256)
void gemm_bf16_wmma(const float* __restrict__ A, const float* __restrict__ B,
                    float* __restrict__ C, int M, int N, int K, int relu)
{
    __shared__ bf16 As[BM * 48];
    __shared__ bf16 Bs[BN * 48];

    const int tid = threadIdx.x, wid = tid >> 5, lane = tid & 31;
    const int lrow = lane & 15, lhalf = lane >> 4;
    const int wm = wid >> 1, wn = wid & 1;
    const int bm0 = blockIdx.y * BM, bn0 = blockIdx.x * BN;

    v8f acc[2][4] = {};
    const int ar = tid >> 3;
    const int ac = (tid & 7) * 4;
    const int acp = ac;

    for (int k0 = 0; k0 < K; k0 += BK) {
#pragma unroll
        for (int i = 0; i < 4; i++) {
            const int r = ar + 32 * i;
            const float4 f = *(const float4*)&A[(size_t)(bm0 + r) * K + k0 + ac];
            bf16* dst = &As[r * 48 + acp];
            dst[0] = (bf16)f.x; dst[1] = (bf16)f.y; dst[2] = (bf16)f.z; dst[3] = (bf16)f.w;
        }
        if (TB) {
#pragma unroll
            for (int i = 0; i < 4; i++) {
                const int n = ar + 32 * i;
                const float4 f = *(const float4*)&B[(size_t)(bn0 + n) * K + k0 + ac];
                bf16* dst = &Bs[n * 48 + ac];
                dst[0] = (bf16)f.x; dst[1] = (bf16)f.y; dst[2] = (bf16)f.z; dst[3] = (bf16)f.w;
            }
        } else {
#pragma unroll
            for (int i = 0; i < 4; i++) {
                const int k = (tid >> 5) + 8 * i;
                const int c4 = (tid & 31) * 4;
                const float4 f = *(const float4*)&B[(size_t)(k0 + k) * N + bn0 + c4];
                Bs[(c4 + 0) * 48 + k] = (bf16)f.x;
                Bs[(c4 + 1) * 48 + k] = (bf16)f.y;
                Bs[(c4 + 2) * 48 + k] = (bf16)f.z;
                Bs[(c4 + 3) * 48 + k] = (bf16)f.w;
            }
        }
        __syncthreads();

        v16bf afr[2], bfr[4];
#pragma unroll
        for (int i = 0; i < 2; i++)
            afr[i] = *(const v16bf*)&As[(wm * 32 + i * 16 + lrow) * 48 + 16 * lhalf];
#pragma unroll
        for (int j = 0; j < 4; j++)
            bfr[j] = *(const v16bf*)&Bs[(wn * 64 + j * 16 + lrow) * 48 + 16 * lhalf];
#pragma unroll
        for (int i = 0; i < 2; i++)
#pragma unroll
            for (int j = 0; j < 4; j++)
                acc[i][j] = __builtin_amdgcn_wmma_f32_16x16x32_bf16(
                    false, afr[i], false, bfr[j], (short)0, acc[i][j], false, false);
        __syncthreads();
    }
#pragma unroll
    for (int i = 0; i < 2; i++)
#pragma unroll
        for (int j = 0; j < 4; j++) {
            const int gm = bm0 + wm * 32 + i * 16 + 8 * lhalf;
            const int gn = bn0 + wn * 64 + j * 16 + lrow;
#pragma unroll
            for (int r = 0; r < 8; r++) {
                float v = acc[i][j][r];
                if (relu) v = fmaxf(v, 0.0f);
                C[(size_t)(gm + r) * N + gn] = v;
            }
        }
}

// ============================ helpers ================================

__global__ __launch_bounds__(256)
void cvt_f32_bf16_kernel(const float* __restrict__ s, bf16* __restrict__ d, long n4)
{
    const long i = (long)blockIdx.x * 256 + threadIdx.x;  // in float4 units
    if (i >= n4) return;
    const float4 f = ((const float4*)s)[i];
    union { bf16 h[4]; uint2 u; } p;
    p.h[0] = (bf16)f.x; p.h[1] = (bf16)f.y; p.h[2] = (bf16)f.z; p.h[3] = (bf16)f.w;
    ((uint2*)d)[i] = p.u;
}

// d[n][k] = (bf16) s[k][n]  (square dim x dim)
__global__ __launch_bounds__(256)
void transpose_cvt_kernel(const float* __restrict__ s, bf16* __restrict__ d, int dim)
{
    const int n = blockIdx.x;
    for (int k = threadIdx.x; k < dim; k += 256)
        d[(size_t)n * dim + k] = (bf16)s[(size_t)k * dim + n];
}

__global__ __launch_bounds__(256)
void rownorm_bf16_kernel(const float* __restrict__ H, bf16* __restrict__ E, int D)
{
    __shared__ float red[256];
    const int row = blockIdx.x;
    const float* h = H + (size_t)row * D;
    float s = 0.0f;
    for (int j = threadIdx.x; j < D; j += 256) { const float v = h[j]; s += v * v; }
    red[threadIdx.x] = s;
    __syncthreads();
    for (int off = 128; off > 0; off >>= 1) {
        if (threadIdx.x < off) red[threadIdx.x] += red[threadIdx.x + off];
        __syncthreads();
    }
    const float inv = 1.0f / fmaxf(sqrtf(red[0]), 1e-12f);
    for (int j = threadIdx.x; j < D; j += 256) E[(size_t)row * D + j] = (bf16)(h[j] * inv);
}

__global__ __launch_bounds__(256)
void rownorm_f32_kernel(const float* __restrict__ H, float* __restrict__ E, int D)
{
    __shared__ float red[256];
    const int row = blockIdx.x;
    const float* h = H + (size_t)row * D;
    float s = 0.0f;
    for (int j = threadIdx.x; j < D; j += 256) { const float v = h[j]; s += v * v; }
    red[threadIdx.x] = s;
    __syncthreads();
    for (int off = 128; off > 0; off >>= 1) {
        if (threadIdx.x < off) red[threadIdx.x] += red[threadIdx.x + off];
        __syncthreads();
    }
    const float inv = 1.0f / fmaxf(sqrtf(red[0]), 1e-12f);
    for (int j = threadIdx.x; j < D; j += 256) E[(size_t)row * D + j] = h[j] * inv;
}

#define NROW 8192
__global__ __launch_bounds__(256)
void topk_mask_kernel(float* __restrict__ S, int N, int KTOP)
{
    __shared__ float vals[NROW];
    __shared__ float rv[256];
    __shared__ int   ri[256];
    __shared__ float topv[64];
    __shared__ int   topi[64];

    const int row = blockIdx.x;
    float* srow = S + (size_t)row * N;
    for (int j = threadIdx.x; j < N; j += 256) vals[j] = srow[j];
    __syncthreads();

    for (int t = 0; t < KTOP; t++) {
        float bv = -__builtin_inff(); int bi = 0x7fffffff;
        for (int j = threadIdx.x; j < N; j += 256) {
            const float v = vals[j];
            if (v > bv) { bv = v; bi = j; }
        }
        rv[threadIdx.x] = bv; ri[threadIdx.x] = bi;
        __syncthreads();
        for (int off = 128; off > 0; off >>= 1) {
            if (threadIdx.x < off) {
                const float ov = rv[threadIdx.x + off]; const int oi = ri[threadIdx.x + off];
                if (ov > rv[threadIdx.x] || (ov == rv[threadIdx.x] && oi < ri[threadIdx.x])) {
                    rv[threadIdx.x] = ov; ri[threadIdx.x] = oi;
                }
            }
            __syncthreads();
        }
        if (threadIdx.x == 0) { topv[t] = rv[0]; topi[t] = ri[0]; vals[ri[0]] = -__builtin_inff(); }
        __syncthreads();
    }
    for (int j = threadIdx.x; j < N; j += 256) {
        float out = 0.0f;
#pragma unroll
        for (int t = 0; t < 31; t++)
            if (t < KTOP && topi[t] == j) out = fmaxf(topv[t], 0.0f);
        srow[j] = out;
    }
}

// ============================ launch ================================

extern "C" void kernel_launch(void* const* d_in, const int* in_sizes, int n_in,
                              void* d_out, int out_size, void* d_ws, size_t ws_size,
                              hipStream_t stream)
{
    const float* features = (const float*)d_in[0];  // [8192,512]
    const float* adj      = (const float*)d_in[1];  // [8192,8192]
    const float* W0       = (const float*)d_in[2];  // [512,512]
    const float* W1       = (const float*)d_in[3];  // [512,512]
    float* out = (float*)d_out;                     // [8192,8192]

    const int N = 8192, D = 512, KTOP = 31;
    dim3 blk(256);
    dim3 gridND(D / BN, N / BM);   // 4 x 64
    dim3 gridNN(N / BN, N / BM);   // 64 x 64

    // primary ws layout (bf16 operands)
    size_t off = 0;
    auto take = [&](size_t bytes) { void* p = (char*)d_ws + off; off += (bytes + 255) & ~(size_t)255; return p; };
    bf16* adjbf = (bf16*)take((size_t)N * N * 2);    // 128 MB
    bf16* fbf   = (bf16*)take((size_t)N * D * 2);    //   8 MB
    bf16* W0t   = (bf16*)take((size_t)D * D * 2);
    bf16* W1t   = (bf16*)take((size_t)D * D * 2);
    bf16* Xt    = (bf16*)take((size_t)D * N * 2);    //   8 MB (X1^T then X2^T)
    bf16* H1    = (bf16*)take((size_t)N * D * 2);    //   8 MB
    float* H2   = (float*)take((size_t)N * D * 4);   //  16 MB
    bf16* embbf = (bf16*)take((size_t)N * D * 2);    //   8 MB
    const size_t need = off;

    if (ws_size >= need) {
        // --- pre-convert operands to bf16 (adj fits CDNA5's 192MB L2 as bf16) ---
        const long adj4 = (long)N * N / 4, f4 = (long)N * D / 4;
        cvt_f32_bf16_kernel<<<dim3((unsigned)((adj4 + 255) / 256)), blk, 0, stream>>>(adj, adjbf, adj4);
        cvt_f32_bf16_kernel<<<dim3((unsigned)((f4 + 255) / 256)), blk, 0, stream>>>(features, fbf, f4);
        transpose_cvt_kernel<<<dim3(D), blk, 0, stream>>>(W0, W0t, D);
        transpose_cvt_kernel<<<dim3(D), blk, 0, stream>>>(W1, W1t, D);

        // 1: X1^T = (features @ W0)^T           [D][N] bf16
        gemm_nt_bf16_async<<<gridND, blk, 0, stream>>>(fbf, W0t, Xt, N, D, D, 2);
        // 2: H1 = relu(adj @ X1)                [N][D] bf16
        gemm_nt_bf16_async<<<gridND, blk, 0, stream>>>(adjbf, Xt, H1, N, D, N, 1);
        // 3: X2^T = (H1 @ W1)^T                 [D][N] bf16
        gemm_nt_bf16_async<<<gridND, blk, 0, stream>>>(H1, W1t, Xt, N, D, D, 2);
        // 4: H2 = adj @ X2                      [N][D] fp32
        gemm_nt_bf16_async<<<gridND, blk, 0, stream>>>(adjbf, Xt, H2, N, D, N, 0);
        // 5: emb = normalize(H2)                [N][D] bf16
        rownorm_bf16_kernel<<<dim3(N), blk, 0, stream>>>(H2, embbf, D);
        // 6: sims = emb @ emb^T -> out          fp32
        gemm_nt_bf16_async<<<gridNN, blk, 0, stream>>>(embbf, embbf, out, N, N, D, 0);
        // 7: top-(K+1) mask + relu, in place
        topk_mask_kernel<<<dim3(N), blk, 0, stream>>>(out, N, KTOP);
    } else {
        // --- fallback: fp32 operands, convert in GEMM staging (needs 32 MB ws) ---
        float* ws0 = (float*)d_ws;
        float* ws1 = ws0 + (size_t)N * D;
        gemm_bf16_wmma<false><<<gridND, blk, 0, stream>>>(features, W0, ws0, N, D, D, 0);
        gemm_bf16_wmma<false><<<gridND, blk, 0, stream>>>(adj, ws0, ws1, N, D, N, 1);
        gemm_bf16_wmma<false><<<gridND, blk, 0, stream>>>(ws1, W1, ws0, N, D, D, 0);
        gemm_bf16_wmma<false><<<gridND, blk, 0, stream>>>(adj, ws0, ws1, N, D, N, 0);
        rownorm_f32_kernel<<<dim3(N), blk, 0, stream>>>(ws1, ws0, D);
        gemm_bf16_wmma<true><<<gridNN, blk, 0, stream>>>(ws0, ws0, out, N, N, D, 0);
        topk_mask_kernel<<<dim3(N), blk, 0, stream>>>(out, N, KTOP);
    }
}
